// TopKSAE_17523466567979
// MI455X (gfx1250) — matrix operand order repaired and checked by reading, not compiled, not run on hardware
//
#include <hip/hip_runtime.h>
#include <stdint.h>

#define B_DIM   65536
#define I_DIM   640
#define L_DIM   2560
#define K_SEL   32

#define KC      64                 // K-chunk staged per TDM transfer
#define NB      128                // latent rows per workgroup tile
#define KCP     72                 // LDS row stride in ushorts (64 + 8 pad via TDM pad)
#define SLAB_B  (NB * KCP * 2)     // bytes per LDS buffer (18432)

#define GLB __attribute__((address_space(1)))
typedef __attribute__((address_space(3))) unsigned short lds_ushort;

typedef __attribute__((ext_vector_type(16))) __bf16 v16bf;
typedef __attribute__((ext_vector_type(8)))  __bf16 v8bf;
typedef __attribute__((ext_vector_type(8)))  float  v8f;
typedef __attribute__((ext_vector_type(4)))  float  f32x4;
typedef __attribute__((ext_vector_type(4)))  unsigned short u16x4;
typedef __attribute__((ext_vector_type(4)))  unsigned int u32x4;
typedef __attribute__((ext_vector_type(8)))  int i32x8;
typedef __attribute__((ext_vector_type(4)))  int i32x4;

// ---------------------------------------------------------------- utilities
__device__ __forceinline__ unsigned short f32_to_bf16_rne(float f) {
    unsigned int u = __builtin_bit_cast(unsigned int, f);
    u = u + 0x7FFFu + ((u >> 16) & 1u);
    return (unsigned short)(u >> 16);
}

__device__ __forceinline__ v16bf cat8(v8bf lo, v8bf hi) {
    return __builtin_shufflevector(lo, hi, 0,1,2,3,4,5,6,7,8,9,10,11,12,13,14,15);
}

// ---------------------------------------------------------- fp32 -> bf16
__global__ void k_cvt_bf16(const float* __restrict__ s,
                           unsigned short* __restrict__ d, int n) {
    int i = (blockIdx.x * blockDim.x + threadIdx.x) * 4;
    if (i < n) {
        f32x4 f = *(const GLB f32x4*)(uintptr_t)(s + i);
        u16x4 o;
        o[0] = f32_to_bf16_rne(f[0]);
        o[1] = f32_to_bf16_rne(f[1]);
        o[2] = f32_to_bf16_rne(f[2]);
        o[3] = f32_to_bf16_rne(f[3]);
        *(GLB u16x4*)(uintptr_t)(d + i) = o;
    }
}

// ------------------------------------------- Wd [I x L] -> WdT [L x I] fp32
__global__ void k_transpose(const float* __restrict__ src,
                            float* __restrict__ dst, int rows, int cols) {
    __shared__ float tile[32][33];
    int c0 = blockIdx.x * 32, r0 = blockIdx.y * 32;
    int tx = threadIdx.x, ty = threadIdx.y;              // (32, 8)
    #pragma unroll
    for (int j = 0; j < 32; j += 8)
        tile[ty + j][tx] =
            ((const GLB float*)(uintptr_t)src)[(size_t)(r0 + ty + j) * cols + (c0 + tx)];
    __syncthreads();
    #pragma unroll
    for (int j = 0; j < 32; j += 8)
        ((GLB float*)(uintptr_t)dst)[(size_t)(c0 + ty + j) * rows + (r0 + tx)] =
            tile[tx][ty + j];
}

// ------------------------------------------- TDM: stage 128 x 64 bf16 slab
// D# per ISA ch.8: group0 {count=1, lds_addr, global_addr, type=2},
// group1 {data_size=2B, pad every 32 dwords by 4 dwords, dims/strides}.
__device__ __forceinline__ void tdm_load_slab(unsigned lds_byte_addr,
                                              const unsigned short* gbase,
                                              unsigned row0, unsigned col0) {
    unsigned long long ga = (unsigned long long)(uintptr_t)gbase +
        ((unsigned long long)row0 * I_DIM + col0) * 2ull;
    u32x4 g0;
    g0[0] = 1u;                                        // count=1 (user D#)
    g0[1] = lds_byte_addr;                             // lds_addr
    g0[2] = (unsigned)(ga & 0xFFFFFFFFu);              // global_addr[31:0]
    g0[3] = (unsigned)((ga >> 32) & 0x01FFFFFFu)       // global_addr[56:32]
          | (2u << 30);                                // type=2 (image)
    i32x8 g1;
    g1[0] = (int)((1u << 16)                           // data_size = 2 bytes
          | (1u << 20)                                 // pad_enable
          | (4u << 22)                                 // pad_interval: 32 dwords
          | (3u << 25));                               // pad_amount: 4 dwords (16B)
    g1[1] = (int)((unsigned)(I_DIM & 0xFFFF) << 16);   // tensor_dim0 lo16
    g1[2] = (int)((unsigned)(L_DIM & 0xFFFF) << 16);   // dim0 hi=0 | tensor_dim1 lo16
    g1[3] = (int)((unsigned)(KC    & 0xFFFF) << 16);   // dim1 hi=0 | tile_dim0=64
    g1[4] = (int)(NB & 0xFFFF);                        // tile_dim1=128, tile_dim2=0
    g1[5] = (int)I_DIM;                                // tensor_dim0_stride lo32
    g1[6] = 0;                                         // stride hi | dim1_stride lo
    g1[7] = 0;
    i32x4 z4 = {0, 0, 0, 0};
    i32x8 z8 = {0, 0, 0, 0, 0, 0, 0, 0};
    __builtin_amdgcn_tensor_load_to_lds(g0, g1, z4, z4, z8, 0);
}

// ------------------------------------------------- encode GEMM (WMMA bf16)
// lat[b, l] = sum_i x[b,i] * We[l,i] + be[l]
// block = 256 thr (8 waves) -> 128(M) x 128(N); wave -> 32(M) x 64(N).
// B (We) slabs are DMA'd into LDS by the TDM, double buffered; A streams
// from L2-resident bf16 x with global_load_b128.
__global__ __launch_bounds__(256)
void k_encode_wmma(const unsigned short* __restrict__ Xb,   // [B_DIM x I_DIM] bf16
                   const unsigned short* __restrict__ Wb,   // [L_DIM x I_DIM] bf16
                   const float* __restrict__ be,
                   float* __restrict__ lat) {               // [B_DIM x L_DIM] f32
    __shared__ unsigned short Bsh[2][NB * KCP];             // 2 x 18 KB (padded rows)

    const int lane = threadIdx.x & 31;
    const int wv   = threadIdx.x >> 5;
    const int wn   = wv & 1;          // 2 waves across N
    const int wm   = wv >> 1;         // 4 waves across M
    const int m0   = blockIdx.y * 128 + wm * 32;
    const int l15  = lane & 15;
    const int hi16 = lane >> 4;       // 0 | 1

    const unsigned lds_base = (unsigned)(size_t)(lds_ushort*)&Bsh[0][0];
    const unsigned nrow0    = blockIdx.x * NB;

    // A-operand (16x32 bf16, ISA 7.12.2): lane m holds K [0..7]+[16..23]
    // (lanes 16..31: +8).
    const GLB unsigned short* ap[2];
    #pragma unroll
    for (int mi = 0; mi < 2; ++mi)
        ap[mi] = (const GLB unsigned short*)(uintptr_t)Xb +
                 (size_t)(m0 + mi * 16 + l15) * I_DIM + hi16 * 8;

    v8f acc[2][4];
    #pragma unroll
    for (int mi = 0; mi < 2; ++mi)
        #pragma unroll
        for (int ni = 0; ni < 4; ++ni)
            acc[mi][ni] = (v8f){0.f,0.f,0.f,0.f,0.f,0.f,0.f,0.f};

    const int NCHUNK = I_DIM / KC;                          // 10

    if (wv == 0)                                            // prologue DMA
        tdm_load_slab(lds_base, Wb, nrow0, 0);

    for (int kc = 0; kc < NCHUNK; ++kc) {
        if (wv == 0) {
            if (kc + 1 < NCHUNK) {
                tdm_load_slab(lds_base + ((kc + 1) & 1) * SLAB_B,
                              Wb, nrow0, (unsigned)(kc + 1) * KC);
                __builtin_amdgcn_s_wait_tensorcnt(1);       // slab kc landed
            } else {
                __builtin_amdgcn_s_wait_tensorcnt(0);
            }
        }
        __syncthreads();                                    // slab kc visible

        const unsigned short* slab = &Bsh[kc & 1][0];
        #pragma unroll
        for (int ks = 0; ks < KC; ks += 32) {
            v16bf a[2], b[4];
            #pragma unroll
            for (int mi = 0; mi < 2; ++mi) {
                a[mi] = cat8(*(const GLB v8bf*)(ap[mi]),
                             *(const GLB v8bf*)(ap[mi] + 16));
                ap[mi] += 32;
            }
            // B-operand (32x16 bf16): lane n holds K [0..7]+[16..23]
            // (lanes 16..31: +8); padded LDS row stride = 72 ushorts.
            #pragma unroll
            for (int ni = 0; ni < 4; ++ni) {
                const unsigned short* bp =
                    slab + (wn * 64 + ni * 16 + l15) * KCP + ks + hi16 * 8;
                b[ni] = cat8(*(const v8bf*)bp, *(const v8bf*)(bp + 16));
            }
            #pragma unroll
            for (int mi = 0; mi < 2; ++mi)
                #pragma unroll
                for (int ni = 0; ni < 4; ++ni)
                    acc[mi][ni] = __builtin_amdgcn_wmma_f32_16x16x32_bf16(
                        false, a[mi], false, b[ni], (short)0, acc[mi][ni],
                        false, false);
        }
        __syncthreads();                                    // done reading slab
    }

    // epilogue: + bias, store dense latents (global stores)
    GLB float* latg = (GLB float*)(uintptr_t)lat;
    const GLB float* beg = (const GLB float*)(uintptr_t)be;
    #pragma unroll
    for (int ni = 0; ni < 4; ++ni) {
        const int ncol = nrow0 + wn * 64 + ni * 16 + l15;
        const float bias = beg[ncol];
        #pragma unroll
        for (int mi = 0; mi < 2; ++mi) {
            const int rbase = m0 + mi * 16 + hi16 * 8;
            v8f c = acc[mi][ni];
            #pragma unroll
            for (int e = 0; e < 8; ++e)
                latg[(size_t)(rbase + e) * L_DIM + ncol] = c[e] + bias;
        }
    }
}

// ------------------------------------------------- per-row top-32 (wave32)
__global__ __launch_bounds__(32)
void k_topk(float* __restrict__ lat, float* __restrict__ topv,
            int* __restrict__ topi) {
    const int r = blockIdx.x;
    const int lane = threadIdx.x;
    GLB float* row = (GLB float*)(uintptr_t)(lat) + (size_t)r * L_DIM;

    float v[80];
    #pragma unroll
    for (int i = 0; i < 80; ++i) v[i] = row[i * 32 + lane];

    uint64_t m0 = 0; uint32_t m1 = 0;
    for (int it = 0; it < K_SEL; ++it) {
        float bv = -__builtin_inff();
        int bs = 0;
        #pragma unroll
        for (int i = 0; i < 80; ++i) {
            bool sel = (i < 64) ? (((m0 >> i) & 1ull) != 0)
                                : (((m1 >> (i - 64)) & 1u) != 0);
            if (!sel && v[i] > bv) { bv = v[i]; bs = i; }
        }
        int bg = bs * 32 + lane;            // global column index
        #pragma unroll
        for (int off = 16; off > 0; off >>= 1) {
            float ov = __shfl_xor(bv, off, 32);
            int   og = __shfl_xor(bg, off, 32);
            if (ov > bv || (ov == bv && og < bg)) { bv = ov; bg = og; }
        }
        if (lane == 0) {
            ((GLB float*)(uintptr_t)topv)[(size_t)r * K_SEL + it] = bv;
            ((GLB int*)(uintptr_t)topi)[(size_t)r * K_SEL + it] = bg;
        }
        if ((bg & 31) == lane) {
            int s = bg >> 5;
            if (s < 64) m0 |= (1ull << s); else m1 |= (1u << (s - 64));
        }
    }

    #pragma unroll
    for (int i = 0; i < 80; ++i) {
        bool sel = (i < 64) ? (((m0 >> i) & 1ull) != 0)
                            : (((m1 >> (i - 64)) & 1u) != 0);
        row[i * 32 + lane] = sel ? v[i] : 0.0f;
    }
}

// ------------------------------------------------------------ sparse decode
__global__ __launch_bounds__(128)
void k_decode(const float* __restrict__ topv, const int* __restrict__ topi,
              const float* __restrict__ WdT, const float* __restrict__ bd,
              float* __restrict__ recon) {
    __shared__ float sv[K_SEL];
    __shared__ int   si[K_SEL];
    const int r = blockIdx.x, t = threadIdx.x;
    if (t < K_SEL) {
        sv[t] = ((const GLB float*)(uintptr_t)topv)[(size_t)r * K_SEL + t];
        si[t] = ((const GLB int*)(uintptr_t)topi)[(size_t)r * K_SEL + t];
    }
    __syncthreads();
    const GLB float* bdg = (const GLB float*)(uintptr_t)bd;
    float acc[5];
    #pragma unroll
    for (int j = 0; j < 5; ++j) acc[j] = bdg[t + j * 128];
    for (int k = 0; k < K_SEL; ++k) {
        const float vv = sv[k];
        const GLB float* wrow =
            (const GLB float*)(uintptr_t)WdT + (size_t)si[k] * I_DIM;
        #pragma unroll
        for (int j = 0; j < 5; ++j) acc[j] += vv * wrow[t + j * 128];
    }
    GLB float* rg = (GLB float*)(uintptr_t)recon + (size_t)r * I_DIM;
    #pragma unroll
    for (int j = 0; j < 5; ++j) rg[t + j * 128] = acc[j];
}

// ---------------------------------------------------------------- launcher
extern "C" void kernel_launch(void* const* d_in, const int* in_sizes, int n_in,
                              void* d_out, int out_size, void* d_ws, size_t ws_size,
                              hipStream_t stream) {
    (void)in_sizes; (void)n_in; (void)out_size; (void)ws_size;
    const float* x  = (const float*)d_in[0];   // [B, I]
    const float* We = (const float*)d_in[1];   // [L, I]
    const float* be = (const float*)d_in[2];   // [L]
    const float* Wd = (const float*)d_in[3];   // [I, L]
    const float* bd = (const float*)d_in[4];   // [I]

    float* recon = (float*)d_out;                               // [B, I]
    float* lat   = recon + (size_t)B_DIM * I_DIM;               // [B, L]

    char* ws = (char*)d_ws;
    unsigned short* Xb  = (unsigned short*)(ws);                // 83,886,080 B
    unsigned short* Wb  = (unsigned short*)(ws +  83886080);    //  3,276,800 B
    float*          WdT = (float*)        (ws +  87162880);     //  6,553,600 B
    float*          tv  = (float*)        (ws +  93716480);     //  8,388,608 B
    int*            ti  = (int*)          (ws + 102105088);     //  8,388,608 B

    { int n = B_DIM * I_DIM;
      k_cvt_bf16<<<n / 4 / 256, 256, 0, stream>>>(x, Xb, n); }
    { int n = L_DIM * I_DIM;
      k_cvt_bf16<<<n / 4 / 256, 256, 0, stream>>>(We, Wb, n); }
    k_transpose<<<dim3(L_DIM / 32, I_DIM / 32), dim3(32, 8), 0, stream>>>(
        Wd, WdT, I_DIM, L_DIM);
    k_encode_wmma<<<dim3(L_DIM / NB, B_DIM / 128), 256, 0, stream>>>(
        Xb, Wb, be, lat);
    k_topk<<<B_DIM, 32, 0, stream>>>(lat, tv, ti);
    k_decode<<<B_DIM, 128, 0, stream>>>(tv, ti, WdT, bd, recon);
}
